// GNNStack_84954453115032
// MI455X (gfx1250) — matrix operand order repaired and checked by reading, not compiled, not run on hardware
//
#include <hip/hip_runtime.h>
#include <math.h>

// ---------------------------------------------------------------------------
// CDNA5 (gfx1250) implementation of the GNNStack reference.
// All dense matmuls go through v_wmma_f32_16x16x32_f16 (wave32 WMMA).
// ---------------------------------------------------------------------------

typedef __attribute__((ext_vector_type(16))) _Float16 v16h;
typedef __attribute__((ext_vector_type(8)))  float    v8f;

struct alignas(16) I4 { unsigned int x, y, z, w; };
union Frag16 { v16h h; I4 q[2]; };

#define HID 256
#define LIN 196
#define KP1 224   // LIN padded to multiple of 32

// ---------------------------------------------------------------------------
// Generic WMMA GEMM: Out[M][Nout] = A[M][Kpad](f16) * B[Kpad][Nout](f16) + bias
// Kpad % 32 == 0, Nout % 64 == 0. One wave computes a 16x64 tile.
// A fragment: lane m = lane&15, K halves at kb..kb+7 and kb+16..kb+23, kb=8*(lane>>4)
// B fragment: lane n = lane&15, K halves at kb..kb+7 and kb+16..kb+23 (same as A)
// ---------------------------------------------------------------------------
__global__ __launch_bounds__(256)
void gemm_wmma_f16(const _Float16* __restrict__ A, const _Float16* __restrict__ B,
                   const float* __restrict__ bias, float* __restrict__ Out,
                   int M, int Kpad, int Nout)
{
    const int ngroups = Nout >> 6;                      // 64-wide column groups
    const int wid  = blockIdx.x * 8 + (threadIdx.x >> 5);
    const int lane = threadIdx.x & 31;
    const int mt = wid / ngroups;
    const int ng = wid % ngroups;
    const int m0 = mt * 16;
    if (m0 >= M) return;
    const int n0 = ng * 64;

    int am = m0 + (lane & 15);
    if (am >= M) am = M - 1;                            // clamp (stores are masked)
    const int kb = (lane >> 4) * 8;
    const _Float16* __restrict__ arow = A + (size_t)am * Kpad + kb;

    const int bk = (lane >> 4) * 8;
    const _Float16* __restrict__ brow = B + (size_t)bk * Nout + n0 + (lane & 15);

    v8f acc0 = {}, acc1 = {}, acc2 = {}, acc3 = {};

    for (int k0 = 0; k0 < Kpad; k0 += 32) {
        Frag16 a;
        a.q[0] = *reinterpret_cast<const I4*>(arow + k0);
        a.q[1] = *reinterpret_cast<const I4*>(arow + k0 + 16);
        const _Float16* bp = brow + (size_t)k0 * Nout;
        Frag16 b;
        for (int s = 0; s < 16; ++s) b.h[s] = bp[(size_t)(s < 8 ? s : s + 8) * Nout + 0];
        acc0 = __builtin_amdgcn_wmma_f32_16x16x32_f16(false, a.h, false, b.h,
                                                      (short)0, acc0, false, false);
        for (int s = 0; s < 16; ++s) b.h[s] = bp[(size_t)(s < 8 ? s : s + 8) * Nout + 16];
        acc1 = __builtin_amdgcn_wmma_f32_16x16x32_f16(false, a.h, false, b.h,
                                                      (short)0, acc1, false, false);
        for (int s = 0; s < 16; ++s) b.h[s] = bp[(size_t)(s < 8 ? s : s + 8) * Nout + 32];
        acc2 = __builtin_amdgcn_wmma_f32_16x16x32_f16(false, a.h, false, b.h,
                                                      (short)0, acc2, false, false);
        for (int s = 0; s < 16; ++s) b.h[s] = bp[(size_t)(s < 8 ? s : s + 8) * Nout + 48];
        acc3 = __builtin_amdgcn_wmma_f32_16x16x32_f16(false, a.h, false, b.h,
                                                      (short)0, acc3, false, false);
    }

    // C/D layout: VGPR r -> M = r + 8*(lane>>4), N = lane&15
    const int cn   = lane & 15;
    const int mbas = (lane >> 4) << 3;
    const float bz0 = bias[n0 + cn +  0];
    const float bz1 = bias[n0 + cn + 16];
    const float bz2 = bias[n0 + cn + 32];
    const float bz3 = bias[n0 + cn + 48];
    #pragma unroll
    for (int r = 0; r < 8; ++r) {
        const int m = m0 + mbas + r;
        if (m < M) {
            float* orow = Out + (size_t)m * Nout + n0 + cn;
            orow[ 0] = acc0[r] + bz0;
            orow[16] = acc1[r] + bz1;
            orow[32] = acc2[r] + bz2;
            orow[48] = acc3[r] + bz3;
        }
    }
}

// ---------------------------------------------------------------------------
// Conv 3x3 valid + ReLU, write f16 activations padded [N][224]
// ---------------------------------------------------------------------------
__global__ __launch_bounds__(256)
void conv_relu_f16(const float* __restrict__ x, const float* __restrict__ w,
                   const float* __restrict__ b, _Float16* __restrict__ h16, int N)
{
    __shared__ float ws[144];
    __shared__ float bs[4];
    const int t = threadIdx.x;
    if (t < 144) ws[t] = w[t];
    if (t < 4)   bs[t] = b[t];
    __syncthreads();
    const int idx = blockIdx.x * 256 + t;
    if (idx >= N * KP1) return;
    const int n = idx / KP1, k = idx % KP1;
    if (k >= LIN) { h16[idx] = (_Float16)0.0f; return; }
    const int oc = k / 49, rr = k % 49, oy = rr / 7, ox = rr % 7;
    const float* xb = x + (size_t)n * 4 * 81;
    float acc = bs[oc];
    #pragma unroll
    for (int ic = 0; ic < 4; ++ic)
        #pragma unroll
        for (int ky = 0; ky < 3; ++ky)
            #pragma unroll
            for (int kx = 0; kx < 3; ++kx)
                acc += xb[ic * 81 + (oy + ky) * 9 + (ox + kx)] *
                       ws[(oc * 4 + ic) * 9 + ky * 3 + kx];
    h16[idx] = (_Float16)fmaxf(acc, 0.0f);
}

// ---------------------------------------------------------------------------
// Weight repacks (f32 [Nout][K] row-major -> f16 [K][Nout] transposed)
// ---------------------------------------------------------------------------
__global__ void pack_w1(const float* __restrict__ wsf, const float* __restrict__ wlf,
                        _Float16* __restrict__ B)
{   // B[k][n], k in [0,224), n in [0,512): cols 0-255 = w_self, 256-511 = w_lin
    const int idx = blockIdx.x * 256 + threadIdx.x;
    if (idx >= KP1 * 512) return;
    const int k = idx / 512, n = idx % 512;
    float v = 0.0f;
    if (k < LIN) v = (n < HID) ? wsf[n * LIN + k] : wlf[(n - HID) * LIN + k];
    B[idx] = (_Float16)v;
}

__global__ void pack_sage(const float* __restrict__ wl, const float* __restrict__ wr,
                          _Float16* __restrict__ B)
{   // B[k][n]: k<256 -> wl[n][k] (agg branch); k>=256 -> wr[n][k-256] (root branch)
    const int idx = blockIdx.x * 256 + threadIdx.x;
    if (idx >= 512 * HID) return;
    const int k = idx / HID, n = idx % HID;
    const float v = (k < HID) ? wl[n * HID + k] : wr[n * HID + (k - HID)];
    B[idx] = (_Float16)v;
}

__global__ void pack_sq(const float* __restrict__ w, _Float16* __restrict__ B, int K, int Nn)
{
    const int idx = blockIdx.x * 256 + threadIdx.x;
    if (idx >= K * Nn) return;
    const int k = idx / Nn, n = idx % Nn;
    B[idx] = (_Float16)w[n * K + k];
}

__global__ void pack_bias(const float* __restrict__ b0, const float* __restrict__ b1,
                          float* __restrict__ dst)
{
    const int idx = blockIdx.x * 256 + threadIdx.x;
    if (idx >= 512) return;
    dst[idx] = (idx < HID) ? b0[idx] : b1[idx - HID];
}

// ---------------------------------------------------------------------------
// Graph machinery
// ---------------------------------------------------------------------------
__global__ void zero_f32(float* __restrict__ p, int n)
{
    const int idx = blockIdx.x * 256 + threadIdx.x;
    if (idx < n) p[idx] = 0.0f;
}

__global__ void deg_kernel(const int* __restrict__ ei, int E, int* __restrict__ deg)
{
    const int e = blockIdx.x * 256 + threadIdx.x;
    if (e >= E) return;
    atomicAdd(&deg[ei[E + e]], 1);
}

__global__ void invdeg_kernel(const int* __restrict__ deg, float* __restrict__ inv, int N)
{
    const int n = blockIdx.x * 256 + threadIdx.x;
    if (n >= N) return;
    const int d = deg[n];
    inv[n] = (d > 0) ? 1.0f / (float)d : 0.0f;
}

__global__ void copy_self(const float* __restrict__ SN, float* __restrict__ xcur, int N)
{
    const int idx = blockIdx.x * 256 + threadIdx.x;
    if (idx >= N * HID) return;
    const int n = idx >> 8, c = idx & 255;
    xcur[idx] = SN[(size_t)n * 512 + c];
}

// xcur[dst] += neigh[src] (self-loops masked); neigh = SN cols 256..511
__global__ void msg_scatter(const int* __restrict__ ei, int E,
                            const float* __restrict__ SN, float* __restrict__ xcur)
{
    const int t = blockIdx.x * 256 + threadIdx.x;
    if (t >= E * 64) return;
    const int e = t >> 6, c = (t & 63) << 2;
    const int s = ei[e], d = ei[E + e];
    if (s == d) return;
    const float4 v = *reinterpret_cast<const float4*>(SN + (size_t)s * 512 + HID + c);
    float* dst = xcur + (size_t)d * HID + c;
    atomicAdd(dst + 0, v.x); atomicAdd(dst + 1, v.y);
    atomicAdd(dst + 2, v.z); atomicAdd(dst + 3, v.w);
}

// agg[dst] += xcur[src] (all edges)
__global__ void sage_scatter(const int* __restrict__ ei, int E,
                             const float* __restrict__ xcur, float* __restrict__ agg)
{
    const int t = blockIdx.x * 256 + threadIdx.x;
    if (t >= E * 64) return;
    const int e = t >> 6, c = (t & 63) << 2;
    const int s = ei[e], d = ei[E + e];
    const float4 v = *reinterpret_cast<const float4*>(xcur + (size_t)s * HID + c);
    float* dst = agg + (size_t)d * HID + c;
    atomicAdd(dst + 0, v.x); atomicAdd(dst + 1, v.y);
    atomicAdd(dst + 2, v.z); atomicAdd(dst + 3, v.w);
}

// acat[n] = [ f16(agg*inv_deg) | f16(xcur) ]   (K = 512 fused SAGE input)
__global__ void prep_acat(const float* __restrict__ agg, const float* __restrict__ inv,
                          const float* __restrict__ xcur, _Float16* __restrict__ acat, int N)
{
    const int idx = blockIdx.x * 256 + threadIdx.x;
    if (idx >= N * HID) return;
    const int n = idx >> 8, c = idx & 255;
    acat[(size_t)n * 512 + c]       = (_Float16)(agg[idx] * inv[n]);
    acat[(size_t)n * 512 + HID + c] = (_Float16)xcur[idx];
}

// Per-node epilogue: emb(copy raw), relu, optional LayerNorm; one wave per node.
__global__ __launch_bounds__(256)
void epilogue(const float* __restrict__ gout, float* __restrict__ xcur,
              const float* __restrict__ lng, const float* __restrict__ lnb,
              int doLN, float* __restrict__ emb, _Float16* __restrict__ x16, int N)
{
    const int wid  = blockIdx.x * 8 + (threadIdx.x >> 5);
    const int lane = threadIdx.x & 31;
    if (wid >= N) return;
    const float* row = gout + (size_t)wid * HID;
    float v[8];
    float s = 0.0f;
    #pragma unroll
    for (int j = 0; j < 8; ++j) {
        const int c = lane + 32 * j;
        const float t = row[c];
        if (emb) emb[(size_t)wid * HID + c] = t;      // emb = pre-activation output
        v[j] = fmaxf(t, 0.0f);
        s += v[j];
    }
    if (doLN) {
        #pragma unroll
        for (int o = 16; o >= 1; o >>= 1) s += __shfl_xor(s, o);
        const float mean = s * (1.0f / 256.0f);
        float var = 0.0f;
        #pragma unroll
        for (int j = 0; j < 8; ++j) { const float d = v[j] - mean; var += d * d; }
        #pragma unroll
        for (int o = 16; o >= 1; o >>= 1) var += __shfl_xor(var, o);
        const float r = rsqrtf(var * (1.0f / 256.0f) + 1e-5f);
        #pragma unroll
        for (int j = 0; j < 8; ++j) {
            const int c = lane + 32 * j;
            v[j] = (v[j] - mean) * r * lng[c] + lnb[c];
        }
    }
    #pragma unroll
    for (int j = 0; j < 8; ++j) {
        const int c = lane + 32 * j;
        xcur[(size_t)wid * HID + c] = v[j];
        if (x16) x16[(size_t)wid * HID + c] = (_Float16)v[j];
    }
}

// logits = p1 @ post_w2.T + b2, then log_softmax over 5 classes
__global__ __launch_bounds__(256)
void logits_kernel(const float* __restrict__ p1, const float* __restrict__ w2,
                   const float* __restrict__ b2, float* __restrict__ out, int N)
{
    __shared__ float sw[5 * HID];
    __shared__ float sb[5];
    for (int i = threadIdx.x; i < 5 * HID; i += 256) sw[i] = w2[i];
    if (threadIdx.x < 5) sb[threadIdx.x] = b2[threadIdx.x];
    __syncthreads();
    const int n = blockIdx.x * 256 + threadIdx.x;
    if (n >= N) return;
    const float* row = p1 + (size_t)n * HID;
    float acc[5];
    #pragma unroll
    for (int o = 0; o < 5; ++o) acc[o] = sb[o];
    for (int c = 0; c < HID; ++c) {
        const float xv = row[c];
        #pragma unroll
        for (int o = 0; o < 5; ++o) acc[o] += xv * sw[o * HID + c];
    }
    float mx = acc[0];
    #pragma unroll
    for (int o = 1; o < 5; ++o) mx = fmaxf(mx, acc[o]);
    float se = 0.0f;
    #pragma unroll
    for (int o = 0; o < 5; ++o) se += expf(acc[o] - mx);
    const float lse = mx + logf(se);
    #pragma unroll
    for (int o = 0; o < 5; ++o) out[(size_t)n * 5 + o] = acc[o] - lse;
}

// ---------------------------------------------------------------------------
// Launcher
// ---------------------------------------------------------------------------
static inline int cdiv(long long a, long long b) { return (int)((a + b - 1) / b); }

extern "C" void kernel_launch(void* const* d_in, const int* in_sizes, int n_in,
                              void* d_out, int out_size, void* d_ws, size_t ws_size,
                              hipStream_t stream)
{
    const float* x        = (const float*)d_in[0];
    const int*   ei       = (const int*)  d_in[1];
    const float* conv_w   = (const float*)d_in[2];
    const float* conv_b   = (const float*)d_in[3];
    const float* w_self   = (const float*)d_in[4];
    const float* b_self   = (const float*)d_in[5];
    const float* w_lin    = (const float*)d_in[6];
    const float* b_lin    = (const float*)d_in[7];
    const float* sage_wl  = (const float*)d_in[8];
    const float* sage_bl  = (const float*)d_in[9];
    const float* sage_wr  = (const float*)d_in[10];
    const float* ln_g     = (const float*)d_in[11];
    const float* ln_b     = (const float*)d_in[12];
    const float* post_w1  = (const float*)d_in[13];
    const float* post_b1  = (const float*)d_in[14];
    const float* post_w2  = (const float*)d_in[15];
    const float* post_b2  = (const float*)d_in[16];
    float* out = (float*)d_out;

    const int N = in_sizes[0] / (4 * 9 * 9);
    const int E = in_sizes[1] / 2;

    // ---- workspace layout -------------------------------------------------
    char* ws = (char*)d_ws;
    size_t off = 0;
    auto alloc = [&](size_t bytes) -> char* {
        char* p = ws + off;
        off += (bytes + 255) & ~(size_t)255;
        return p;
    };
    _Float16* wbuf1 = (_Float16*)alloc((size_t)KP1 * 512 * 2);       // first-layer weights
    _Float16* wsg   = (_Float16*)alloc((size_t)3 * 512 * HID * 2);   // 3 fused SAGE weights
    _Float16* wpost = (_Float16*)alloc((size_t)HID * HID * 2);       // post_w1
    float*    bcat  = (float*)   alloc(512 * 4);                     // [b_self|b_lin]
    _Float16* h16   = (_Float16*)alloc((size_t)N * KP1 * 2);         // conv activations
    float*    SN    = (float*)   alloc((size_t)N * 512 * 4);         // [self|neigh]; reused
    float*    xcur  = (float*)   alloc((size_t)N * HID * 4);
    float*    agg   = (float*)   alloc((size_t)N * HID * 4);
    _Float16* acat  = (_Float16*)alloc((size_t)N * 512 * 2);         // [agg|x] f16
    int*      deg   = (int*)     alloc((size_t)N * 4);
    float*    invd  = (float*)   alloc((size_t)N * 4);
    // reuse of SN after the message pass:
    float*    gout  = SN;                                            // GEMM output [N][256]
    _Float16* x16   = (_Float16*)(SN + (size_t)N * HID);             // final acts f16

    // ---- weight packing (tiny) -------------------------------------------
    pack_w1  <<<cdiv((long long)KP1 * 512, 256), 256, 0, stream>>>(w_self, w_lin, wbuf1);
    for (int i = 0; i < 3; ++i)
        pack_sage<<<cdiv(512 * HID, 256), 256, 0, stream>>>(
            sage_wl + (size_t)i * HID * HID, sage_wr + (size_t)i * HID * HID,
            wsg + (size_t)i * 512 * HID);
    pack_sq  <<<cdiv(HID * HID, 256), 256, 0, stream>>>(post_w1, wpost, HID, HID);
    pack_bias<<<2, 256, 0, stream>>>(b_self, b_lin, bcat);

    // ---- conv + relu -> f16 ----------------------------------------------
    conv_relu_f16<<<cdiv((long long)N * KP1, 256), 256, 0, stream>>>(x, conv_w, conv_b, h16, N);

    // ---- GEMM1: [N,224] x [224,512] -> SN = [self_x | neigh] --------------
    {
        const int Mt = cdiv(N, 16), ngroups = 512 / 64;
        gemm_wmma_f16<<<cdiv((long long)Mt * ngroups, 8), 256, 0, stream>>>(
            h16, wbuf1, bcat, SN, N, KP1, 512);
    }

    // ---- degree + CustomConv message pass ---------------------------------
    zero_f32  <<<cdiv(N, 256), 256, 0, stream>>>((float*)deg, N);
    deg_kernel<<<cdiv(E, 256), 256, 0, stream>>>(ei, E, deg);
    invdeg_kernel<<<cdiv(N, 256), 256, 0, stream>>>(deg, invd, N);
    copy_self <<<cdiv((long long)N * HID, 256), 256, 0, stream>>>(SN, xcur, N);
    msg_scatter<<<cdiv((long long)E * 64, 256), 256, 0, stream>>>(ei, E, SN, xcur);

    // ---- 3 x SAGEConv -----------------------------------------------------
    const int Mt = cdiv(N, 16);
    for (int i = 0; i < 3; ++i) {
        zero_f32    <<<cdiv((long long)N * HID, 256), 256, 0, stream>>>(agg, N * HID);
        sage_scatter<<<cdiv((long long)E * 64, 256), 256, 0, stream>>>(ei, E, xcur, agg);
        prep_acat   <<<cdiv((long long)N * HID, 256), 256, 0, stream>>>(agg, invd, xcur, acat, N);
        gemm_wmma_f16<<<cdiv((long long)Mt * 4, 8), 256, 0, stream>>>(
            acat, wsg + (size_t)i * 512 * HID, sage_bl + (size_t)i * HID, gout, N, 512, HID);
        epilogue<<<cdiv(N, 8), 256, 0, stream>>>(
            gout, xcur,
            ln_g + (size_t)(i < 2 ? i : 0) * HID, ln_b + (size_t)(i < 2 ? i : 0) * HID,
            (i < 2) ? 1 : 0,
            (i == 2) ? out : nullptr,            // emb -> first N*256 of d_out
            (i == 2) ? x16 : nullptr, N);
    }

    // ---- post_mp: [N,256] x [256,256] + b1, then 5-way head + log_softmax --
    gemm_wmma_f16<<<cdiv((long long)Mt * 4, 8), 256, 0, stream>>>(
        x16, wpost, post_b1, gout, N, HID, HID);
    logits_kernel<<<cdiv(N, 256), 256, 0, stream>>>(
        gout, post_w2, post_b2, out + (size_t)N * HID, N);
}
